// PowerConsumptionLSTM_16260746183208
// MI455X (gfx1250) — matrix operand order repaired and checked by reading, not compiled, not run on hardware
//
#include <hip/hip_runtime.h>
#include <hip/hip_bf16.h>

// LSTM sizes (match reference)
#define H       100
#define G_REAL  400        // 4*H gates
#define G       416        // padded to 26 x 16 (fake units produce exact h=0)
#define TSTEPS  512
#define BATCH   2048

// Tiling
#define BT      32         // batch rows per workgroup -> 64 workgroups
#define KC0     4          // layer0 K chunks (reads A1 cols 0..127; Bm0 rows 100..127 zero)
#define K0P     128        // Bm0 K extent
#define K1P     224        // layer1 K ([h0;h1]=200, halves padded to 112 -> 7 x 32)
#define KC1     7
#define NTPW    2          // N-tiles (16 cols) per wave per GEMM
#define WAVES   26         // 2 M-tiles x 13 wave-groups x (2 N-tiles each)
#define NTHREADS (WAVES*32)

typedef _Float16 v16h __attribute__((ext_vector_type(16)));
typedef _Float16 v8h  __attribute__((ext_vector_type(8)));
typedef float    v8f  __attribute__((ext_vector_type(8)));

__device__ __forceinline__ float fast_sigmoid(float x) {
  return __builtin_amdgcn_rcpf(1.0f + __expf(-x));
}

// 16-bit A fragment (16M x 32K), CDNA5 layout:
// lanes 0-15: M=lane, K = {0..7} U {16..23}; lanes 16-31: M=lane-16, K = {8..15} U {24..31}
__device__ __forceinline__ v16h load_frag_A(const _Float16* base, int stride,
                                            int m, int kc, int lane) {
  const int k0 = kc * 32 + ((lane >= 16) ? 8 : 0);
  const _Float16* p = base + m * stride + k0;
  v8h lo = *(const v8h*)(p);
  v8h hi = *(const v8h*)(p + 16);
  v16h r;
#pragma unroll
  for (int i = 0; i < 8; i++) { r[i] = lo[i]; r[8 + i] = hi[i]; }
  return r;
}

// 16-bit B fragment (32K x 16N), weights stored transposed in LDS as [col][K]:
// lanes 0-15: N=lane, K = {0..7} U {16..23}; lanes 16-31: N=lane-16, K = {8..15} U {24..31}
__device__ __forceinline__ v16h load_frag_B(const _Float16* base, int stride,
                                            int col, int kc, int lane) {
  const int k0 = kc * 32 + ((lane >= 16) ? 8 : 0);
  const _Float16* p = base + col * stride + k0;
  v8h lo = *(const v8h*)(p);
  v8h hi = *(const v8h*)(p + 16);
  v16h r;
#pragma unroll
  for (int i = 0; i < 8; i++) { r[i] = lo[i]; r[8 + i] = hi[i]; }
  return r;
}

__global__ __launch_bounds__(NTHREADS)
void lstm2_persistent_kernel(const float* __restrict__ x,
                             const float* __restrict__ W_ih0, const float* __restrict__ W_hh0,
                             const float* __restrict__ b_ih0, const float* __restrict__ b_hh0,
                             const float* __restrict__ W_ih1, const float* __restrict__ W_hh1,
                             const float* __restrict__ b_ih1, const float* __restrict__ b_hh1,
                             const float* __restrict__ W_out, const float* __restrict__ b_out,
                             float* __restrict__ out)
{
  extern __shared__ _Float16 smem[];
  _Float16* Bm0 = smem;                 // [G][K0P]  W_hh0^T, gate-interleaved cols, f16 (rows>=100 zero)
  _Float16* Bm1 = Bm0 + G * K0P;        // [G][K1P]  [W_ih1;W_hh1]^T, gate-interleaved
  _Float16* A1  = Bm1 + G * K1P;        // [BT][K1P] cols 0..99: h0, 112..211: h1, rest zero
  float*    xs  = (float*)(A1 + BT * K1P); // [BT][3] staged input for current t

  const int tid  = threadIdx.x;
  const int lane = tid & 31;
  const int wave = tid >> 5;
  const int B0   = blockIdx.x * BT;

  // ---- one-time: stage weights to LDS (f16, transposed, gate-interleaved) ----
  // interleaved column n -> original gate row r = (n&3)*H + (n>>2); n>=400 -> zero
  for (int idx = tid; idx < G * K0P; idx += NTHREADS) {
    int n = idx / K0P, k = idx % K0P;
    int r = (n & 3) * H + (n >> 2);
    float v = (n < G_REAL && k < H) ? W_hh0[r * H + k] : 0.0f;
    Bm0[idx] = (_Float16)v;
  }
  for (int idx = tid; idx < G * K1P; idx += NTHREADS) {
    int n = idx / K1P, k = idx % K1P;
    int r = (n & 3) * H + (n >> 2);
    float v = 0.0f;
    if (n < G_REAL) {
      if (k < H)                      v = W_ih1[r * H + k];
      else if (k >= 112 && k < 112+H) v = W_hh1[r * H + (k - 112)];
    }
    Bm1[idx] = (_Float16)v;
  }
  for (int idx = tid; idx < BT * K1P; idx += NTHREADS) A1[idx] = (_Float16)0.0f;

  // ---- per-wave tile assignment: wave -> (mt, 2 consecutive N-tiles) ----
  const int mt  = wave / 13;            // M tile (0..1): rows mt*16 .. mt*16+15
  const int ntb = (wave % 13) * NTPW;   // first N tile of this wave (0..24, step 2)
  const int lcol = lane & 15;
  const int am  = mt * 16 + lcol;                  // A-fragment row
  const int crl = mt * 16 + ((lane >> 4) << 3);    // C-fragment local row base
  const int src = lane & ~3;                       // base lane of i/f/g/o group
  // unified activation: gate lane g==2 needs tanh(x)=2*sigmoid(2x)-1, others sigmoid(x)
  const bool isg = (lane & 3) == 2;
  const float actS = isg ? 2.0f : 1.0f;
  const float actB = isg ? -1.0f : 0.0f;

  // persistent per-lane constants (column = (ntb+i)*16 + lane%16); fake cols -> 0
  float bias0v[NTPW], bias1v[NTPW], wx0[NTPW], wx1[NTPW], wx2[NTPW];
#pragma unroll
  for (int i = 0; i < NTPW; i++) {
    int col = (ntb + i) * 16 + lcol;
    if (col < G_REAL) {
      int r = (col & 3) * H + (col >> 2);
      bias0v[i] = b_ih0[r] + b_hh0[r];
      bias1v[i] = b_ih1[r] + b_hh1[r];
      wx0[i] = W_ih0[r * 3 + 0];
      wx1[i] = W_ih0[r * 3 + 1];
      wx2[i] = W_ih0[r * 3 + 2];
    } else {
      bias0v[i] = 0.0f; bias1v[i] = 0.0f;
      wx0[i] = 0.0f; wx1[i] = 0.0f; wx2[i] = 0.0f;
    }
  }

  // persistent cell state (f32, registers; only writer lanes (lane&3)==0 use it)
  v8f c0[NTPW], c1[NTPW];
#pragma unroll
  for (int i = 0; i < NTPW; i++) {
#pragma unroll
    for (int v = 0; v < 8; v++) { c0[i][v] = 0.0f; c1[i][v] = 0.0f; }
  }

  __syncthreads();

  for (int t = 0; t < TSTEPS; t++) {
    // stage x[:, t, :] for this block (tiny) + prefetch next step
    for (int idx = tid; idx < BT * 3; idx += NTHREADS) {
      int row = idx / 3, j = idx % 3;
      const float* px = &x[(B0 + row) * (TSTEPS * 3) + t * 3 + j];
      xs[idx] = *px;
      if (t + 1 < TSTEPS) __builtin_prefetch(px + 3, 0, 0);
    }
    __syncthreads();  // prev-step h1 writes + xs visible

    // hoisted per-row x values (shared across the wave's N-tiles)
    float xv0[8], xv1[8], xv2[8];
#pragma unroll
    for (int v = 0; v < 8; v++) {
      int row = crl + v;
      xv0[v] = xs[row * 3 + 0];
      xv1[v] = xs[row * 3 + 1];
      xv2[v] = xs[row * 3 + 2];
    }

    // ---------------- layer 0: gates = bias + x*W_ih0 + h0 @ W_hh0^T ----------------
    // A-matrix = A1 cols 0..127 (cols 100..127 hit zero rows of Bm0)
    v8f g0[NTPW];
#pragma unroll
    for (int i = 0; i < NTPW; i++) {
#pragma unroll
      for (int v = 0; v < 8; v++)
        g0[i][v] = fmaf(wx2[i], xv2[v], fmaf(wx1[i], xv1[v], fmaf(wx0[i], xv0[v], bias0v[i])));
    }
    {
      // software-pipelined fragment loads (distinct regs -> overlap DS latency with WMMA)
      v16h a_cur = load_frag_A(A1, K1P, am, 0, lane);
      v16h b_cur = load_frag_B(Bm0, K0P, ntb * 16 + lcol, 0, lane);
#pragma unroll
      for (int kc = 0; kc < KC0; kc++) {
        v16h a_nxt = a_cur;
        if (kc + 1 < KC0) a_nxt = load_frag_A(A1, K1P, am, kc + 1, lane);
#pragma unroll
        for (int i = 0; i < NTPW; i++) {
          v16h b_nxt = b_cur;
          if (i + 1 < NTPW)      b_nxt = load_frag_B(Bm0, K0P, (ntb+i+1)*16 + lcol, kc, lane);
          else if (kc + 1 < KC0) b_nxt = load_frag_B(Bm0, K0P, ntb*16 + lcol, kc + 1, lane);
          g0[i] = __builtin_amdgcn_wmma_f32_16x16x32_f16(false, a_cur, false, b_cur,
                                                         (short)0, g0[i], false, false);
          b_cur = b_nxt;
        }
        a_cur = a_nxt;
      }
    }
    __syncthreads();  // all A1 reads complete before rewriting h0

    // ---------------- layer 0 cell update (gate-interleaved tiles) ----------------
#pragma unroll
    for (int i = 0; i < NTPW; i++) {
      int ug = (ntb + i) * 4 + (lcol >> 2);  // hidden unit of this lane group (incl. fake pads)
#pragma unroll
      for (int v = 0; v < 8; v++) {
        float gv  = g0[i][v];
        float act = fmaf(actS, fast_sigmoid(actS * gv), actB);  // per-lane gate activation
        float ii = __shfl(act, src + 0, 32);
        float ff = __shfl(act, src + 1, 32);
        float tg = __shfl(act, src + 2, 32);
        float oo = __shfl(act, src + 3, 32);
        if ((lane & 3) == 0) {
          float c = fmaf(ff, c0[i][v], ii * tg);
          c0[i][v] = c;
          float tc = fmaf(2.0f, fast_sigmoid(2.0f * c), -1.0f);  // tanh(c)
          float h  = oo * tc;
          A1[(crl + v) * K1P + ug] = (_Float16)h;   // h0 -> layer-1 A (first half)
        }
      }
    }
    __syncthreads();  // h0 visible

    // ---------------- layer 1: gates = bias + [h0_t;h1_prev] @ [W_ih1;W_hh1]^T ----------------
    v8f g1[NTPW];
#pragma unroll
    for (int i = 0; i < NTPW; i++) {
#pragma unroll
      for (int v = 0; v < 8; v++) g1[i][v] = bias1v[i];
    }
    {
      v16h a_cur = load_frag_A(A1, K1P, am, 0, lane);
      v16h b_cur = load_frag_B(Bm1, K1P, ntb * 16 + lcol, 0, lane);
#pragma unroll
      for (int kc = 0; kc < KC1; kc++) {
        v16h a_nxt = a_cur;
        if (kc + 1 < KC1) a_nxt = load_frag_A(A1, K1P, am, kc + 1, lane);
#pragma unroll
        for (int i = 0; i < NTPW; i++) {
          v16h b_nxt = b_cur;
          if (i + 1 < NTPW)      b_nxt = load_frag_B(Bm1, K1P, (ntb+i+1)*16 + lcol, kc, lane);
          else if (kc + 1 < KC1) b_nxt = load_frag_B(Bm1, K1P, ntb*16 + lcol, kc + 1, lane);
          g1[i] = __builtin_amdgcn_wmma_f32_16x16x32_f16(false, a_cur, false, b_cur,
                                                         (short)0, g1[i], false, false);
          b_cur = b_nxt;
        }
        a_cur = a_nxt;
      }
    }
    __syncthreads();  // all A1 reads complete before rewriting h1

    // ---------------- layer 1 cell update ----------------
#pragma unroll
    for (int i = 0; i < NTPW; i++) {
      int ug = (ntb + i) * 4 + (lcol >> 2);
#pragma unroll
      for (int v = 0; v < 8; v++) {
        float gv  = g1[i][v];
        float act = fmaf(actS, fast_sigmoid(actS * gv), actB);
        float ii = __shfl(act, src + 0, 32);
        float ff = __shfl(act, src + 1, 32);
        float tg = __shfl(act, src + 2, 32);
        float oo = __shfl(act, src + 3, 32);
        if ((lane & 3) == 0) {
          float c = fmaf(ff, c1[i][v], ii * tg);
          c1[i][v] = c;
          float tc = fmaf(2.0f, fast_sigmoid(2.0f * c), -1.0f);
          float h  = oo * tc;
          A1[(crl + v) * K1P + 112 + ug] = (_Float16)h;  // h1 (second half)
        }
      }
    }
    __syncthreads();
  }

  // ---- head: out[b] = h1_last[b,:] . W_out + b_out ----
  if (tid < BT) {
    float acc = b_out[0];
#pragma unroll 4
    for (int u = 0; u < H; u++)
      acc = fmaf((float)A1[tid * K1P + 112 + u], W_out[u], acc);
    out[B0 + tid] = acc;
  }
}

extern "C" void kernel_launch(void* const* d_in, const int* in_sizes, int n_in,
                              void* d_out, int out_size, void* d_ws, size_t ws_size,
                              hipStream_t stream) {
  const float* x     = (const float*)d_in[0];
  const float* W_ih0 = (const float*)d_in[1];
  const float* W_hh0 = (const float*)d_in[2];
  const float* b_ih0 = (const float*)d_in[3];
  const float* b_hh0 = (const float*)d_in[4];
  const float* W_ih1 = (const float*)d_in[5];
  const float* W_hh1 = (const float*)d_in[6];
  const float* b_ih1 = (const float*)d_in[7];
  const float* b_hh1 = (const float*)d_in[8];
  const float* W_out = (const float*)d_in[9];
  const float* b_out = (const float*)d_in[10];
  float* out = (float*)d_out;

  const size_t smem_bytes =
      (size_t)(G * K0P + G * K1P + BT * K1P) * sizeof(_Float16)
      + (size_t)BT * 3 * sizeof(float);   // 307,584 bytes < 320KB WGP LDS

  lstm2_persistent_kernel<<<BATCH / BT, NTHREADS, smem_bytes, stream>>>(
      x, W_ih0, W_hh0, b_ih0, b_hh0,
      W_ih1, W_hh1, b_ih1, b_hh1,
      W_out, b_out, out);
}
